// Model_39676907887322
// MI455X (gfx1250) — matrix operand order repaired and checked by reading, not compiled, not run on hardware
//
#include <hip/hip_runtime.h>
#include <stdint.h>

typedef __attribute__((ext_vector_type(16))) __bf16 v16bf;
typedef __attribute__((ext_vector_type(8)))  __bf16 v8bf;
typedef __attribute__((ext_vector_type(4)))  __bf16 v4bf;
typedef __attribute__((ext_vector_type(8)))  float  v8f;

#define NB 4
#define NH 16
#define NS 2048
#define ND 64
#define QTILE 128          // q rows per block: 8 waves x 16 rows
#define KTILE 64           // keys staged in LDS & softmax granularity
#define LOG2E 1.44269504088896340736f

// DPP16 lane move (stays within a row of 16 lanes; wave32 rows = the two
// WMMA lane-groups). CTRL: 0xB1 = quad_perm xor1, 0x4E = quad_perm xor2,
// 0x141 = row_half_mirror, 0x140 = row_mirror.
template <int CTRL>
__device__ __forceinline__ float dppmove(float x) {
  int xi = __builtin_bit_cast(int, x);
  int yi = __builtin_amdgcn_update_dpp(0, xi, CTRL, 0xf, 0xf, true);
  return __builtin_bit_cast(float, yi);
}

__device__ __forceinline__ float rowmax16(float x) {
  x = fmaxf(x, dppmove<0xB1>(x));
  x = fmaxf(x, dppmove<0x4E>(x));
  x = fmaxf(x, dppmove<0x141>(x));
  x = fmaxf(x, dppmove<0x140>(x));
  return x;
}

__device__ __forceinline__ float rowsum16(float x) {
  x += dppmove<0xB1>(x);
  x += dppmove<0x4E>(x);
  x += dppmove<0x141>(x);
  x += dppmove<0x140>(x);
  return x;
}

// load a 16-element bf16 fragment run-pair (two aligned 16B LDS loads)
__device__ __forceinline__ v16bf ldfrag(const __bf16* p) {
  v8bf lo = *(const v8bf*)(p);
  v8bf hi = *(const v8bf*)(p + 16);
  return __builtin_shufflevector(lo, hi, 0,1,2,3,4,5,6,7,8,9,10,11,12,13,14,15);
}

__device__ __forceinline__ uint32_t mix32(uint32_t x) {
  x ^= x >> 16; x *= 0x7feb352du;
  x ^= x >> 15; x *= 0x846ca68bu;
  x ^= x >> 16; return x;
}

__global__ __launch_bounds__(256)
void fattn_fwd_kernel(const float* __restrict__ Q, const float* __restrict__ K,
                      const float* __restrict__ V, float* __restrict__ O) {
  // K tile row-major [key][d], V tile transposed [d][key], P per-wave [q][key];
  // stride 72 bf16 (144B): 16B-aligned runs, conflict-free b128 fragment loads.
  __shared__ __bf16 sK [KTILE][72];
  __shared__ __bf16 sVt[ND   ][72];
  __shared__ __bf16 sP [8][16][72];

  const int lane  = threadIdx.x & 31;
  const int wv    = threadIdx.x >> 5;
  const int lh    = lane >> 4;      // lane half (WMMA lane-group)
  const int mm    = lane & 15;
  const int bh    = blockIdx.z * NH + blockIdx.y;
  const int qrow0 = blockIdx.x * QTILE + wv * 16;

  const size_t headoff = (size_t)bh * NS * ND;
  const float* Qh = Q + headoff;
  const float* Kh = K + headoff;
  const float* Vh = V + headoff;
  float*       Oh = O + headoff;

  const uint32_t seed  = mix32((uint32_t)bh * 0x9e3779b9u + 0x85ebca6bu);
  const uint32_t bitlo = 1u << mm;          // per-lane dropout bit masks
  const uint32_t bithi = bitlo << 16;

  // ---- this wave's 16x64 Q tile as two bf16 A-fragments (pinned in VGPRs) ----
  // A-layout (16-bit, 16x32): runs at d0 = f*32 + lh*8 and d0 + 16.
  v16bf qf[2];
  {
    const float* qp = Qh + (size_t)(qrow0 + mm) * ND;
#pragma unroll
    for (int f = 0; f < 2; ++f) {
      const int d0 = f * 32 + lh * 8;
      float4 a0 = *(const float4*)(qp + d0);
      float4 a1 = *(const float4*)(qp + d0 + 4);
      float4 a2 = *(const float4*)(qp + d0 + 16);
      float4 a3 = *(const float4*)(qp + d0 + 20);
      v16bf t;
      t[0] = (__bf16)a0.x; t[1] = (__bf16)a0.y; t[2]  = (__bf16)a0.z; t[3]  = (__bf16)a0.w;
      t[4] = (__bf16)a1.x; t[5] = (__bf16)a1.y; t[6]  = (__bf16)a1.z; t[7]  = (__bf16)a1.w;
      t[8] = (__bf16)a2.x; t[9] = (__bf16)a2.y; t[10] = (__bf16)a2.z; t[11] = (__bf16)a2.w;
      t[12]= (__bf16)a3.x; t[13]= (__bf16)a3.y; t[14] = (__bf16)a3.z; t[15] = (__bf16)a3.w;
      qf[f] = t;
    }
  }

  v8f acc[4] = {};                       // 16x64 output accumulator (C layout)
  float mrun[8], lrun[8];                // per-row running max / sum (row = lh*8+r)
#pragma unroll
  for (int r = 0; r < 8; ++r) { mrun[r] = -3.0e38f; lrun[r] = 0.0f; }

  for (int kb = 0; kb < NS; kb += KTILE) {
    // ---- stage 64 keys: K row-major, V transposed, fp32 -> bf16 ----
#pragma unroll
    for (int i = 0; i < 4; ++i) {
      const int idx = threadIdx.x + i * 256;   // 0..1023 float4s
      const int row = idx >> 4;                // 0..63
      const int c4  = (idx & 15) << 2;         // 0..60
      float4 kk = *(const float4*)(Kh + (size_t)(kb + row) * ND + c4);
      v4bf kp; kp[0] = (__bf16)kk.x; kp[1] = (__bf16)kk.y;
               kp[2] = (__bf16)kk.z; kp[3] = (__bf16)kk.w;
      *(v4bf*)&sK[row][c4] = kp;
      float4 vv = *(const float4*)(Vh + (size_t)(kb + row) * ND + c4);
      sVt[c4 + 0][row] = (__bf16)vv.x;
      sVt[c4 + 1][row] = (__bf16)vv.y;
      sVt[c4 + 2][row] = (__bf16)vv.z;
      sVt[c4 + 3][row] = (__bf16)vv.w;
      if (kb + KTILE < NS) {                   // global_prefetch_b8 for next tile
        __builtin_prefetch(Kh + (size_t)(kb + KTILE + row) * ND + c4, 0, 0);
        __builtin_prefetch(Vh + (size_t)(kb + KTILE + row) * ND + c4, 0, 0);
      }
    }
    __syncthreads();

    // ---- S = Q K^T : four 16x16 score tiles (64 keys), 8 WMMAs ----
    // B-layout (16-bit, 32x16): lane-group lh holds K runs at lh*8 and 16 + lh*8, col N = mm.
    v8f s[4];
#pragma unroll
    for (int t = 0; t < 4; ++t) {
      v16bf b0 = ldfrag(&sK[t * 16 + mm][     lh * 8]);
      v16bf b1 = ldfrag(&sK[t * 16 + mm][32 + lh * 8]);
      v8f z = {};
      z = __builtin_amdgcn_wmma_f32_16x16x32_bf16(false, qf[0], false, b0, (short)0, z, false, false);
      z = __builtin_amdgcn_wmma_f32_16x16x32_bf16(false, qf[1], false, b1, (short)0, z, false, false);
      s[t] = z;
    }

    // ---- online softmax over 64 keys (one butterfly per stage) ----
    float mloc[8];
    bool up = false;
#pragma unroll
    for (int r = 0; r < 8; ++r) {
      mloc[r] = rowmax16(fmaxf(fmaxf(s[0][r], s[1][r]), fmaxf(s[2][r], s[3][r])));
      up = up || (mloc[r] > mrun[r]);
    }
    if (__any(up)) {
      // running max grew for some row in this wave: exact rescale path
#pragma unroll
      for (int r = 0; r < 8; ++r) {
        const float mn    = fmaxf(mrun[r], mloc[r]);
        const float alpha = __builtin_exp2f((mrun[r] - mn) * LOG2E);
        mrun[r] = mn;
        lrun[r] *= alpha;
#pragma unroll
        for (int t = 0; t < 4; ++t) acc[t][r] *= alpha;
      }
    } // else: alpha == 1 exactly for every row; skip all rescaling work

    float lsum[8];
#pragma unroll
    for (int r = 0; r < 8; ++r) {
#pragma unroll
      for (int t = 0; t < 4; ++t)
        s[t][r] = __builtin_exp2f((s[t][r] - mrun[r]) * LOG2E);
      lsum[r] = rowsum16((s[0][r] + s[1][r]) + (s[2][r] + s[3][r]));
      lrun[r] += lsum[r];
    }

    // ---- dropout (p=0.5, x2 on numerator) + pack P to LDS (C->A layout) ----
    // One 32-bit hash per (row, 32-key chunk); lane tests its precomputed bit.
#pragma unroll
    for (int r = 0; r < 8; ++r) {
      const int qg = qrow0 + lh * 8 + r;
      const uint32_t base = seed ^ ((uint32_t)qg << 11) ^ (uint32_t)kb;
      const uint32_t h0 = mix32(base);
      const uint32_t h1 = mix32(base + 0x68bc21ebu);
#pragma unroll
      for (int t = 0; t < 4; ++t) {
        const uint32_t h    = (t < 2) ? h0 : h1;
        const uint32_t keep = h & ((t & 1) ? bithi : bitlo);
        const float pv = keep ? s[t][r] * 2.0f : 0.0f;
        sP[wv][lh * 8 + r][t * 16 + mm] = (__bf16)pv;
      }
    }
    asm volatile("s_wait_dscnt 0" ::: "memory");  // per-wave P store->load order

    // ---- O += P V : two 16x32 P A-frags, eight 32x16 V B-frags, 8 WMMAs ----
    v16bf pa[2];
#pragma unroll
    for (int c = 0; c < 2; ++c) {
      v8bf lo = *(const v8bf*)&sP[wv][mm][c * 32 +      lh * 8];
      v8bf hi = *(const v8bf*)&sP[wv][mm][c * 32 + 16 + lh * 8];
      pa[c] = __builtin_shufflevector(lo, hi, 0,1,2,3,4,5,6,7,8,9,10,11,12,13,14,15);
    }
#pragma unroll
    for (int t = 0; t < 4; ++t) {
#pragma unroll
      for (int c = 0; c < 2; ++c) {
        v16bf vb = ldfrag(&sVt[t * 16 + mm][c * 32 + lh * 8]);
        acc[t] = __builtin_amdgcn_wmma_f32_16x16x32_bf16(false, pa[c], false, vb, (short)0, acc[t], false, false);
      }
    }
    __syncthreads();
  }

  // ---- epilogue: normalize by running sum, write fp32 ----
  float invl[8];
#pragma unroll
  for (int r = 0; r < 8; ++r) invl[r] = 1.0f / lrun[r];
#pragma unroll
  for (int t = 0; t < 4; ++t)
#pragma unroll
    for (int r = 0; r < 8; ++r)
      Oh[(size_t)(qrow0 + lh * 8 + r) * ND + t * 16 + mm] = acc[t][r] * invl[r];
}

extern "C" void kernel_launch(void* const* d_in, const int* in_sizes, int n_in,
                              void* d_out, int out_size, void* d_ws, size_t ws_size,
                              hipStream_t stream) {
  (void)in_sizes; (void)n_in; (void)d_ws; (void)ws_size; (void)out_size;
  const float* q = (const float*)d_in[0];
  const float* k = (const float*)d_in[1];
  const float* v = (const float*)d_in[2];
  float* o = (float*)d_out;
  dim3 grid(NS / QTILE, NH, NB);   // (16, 16, 4) = 1024 blocks
  dim3 block(256);                 // 8 wave32s
  fattn_fwd_kernel<<<grid, block, 0, stream>>>(q, k, v, o);
}
